// CorrectAndSmooth_6631429505298
// MI455X (gfx1250) — compile-verified, matrix operand order checked by reading, not run on hardware
//
#include <hip/hip_runtime.h>
#include <stdint.h>
#include <math.h>

#define THREADS 256
#define TILE    1024          // edges staged per block
#define CNUM    40            // classes

// ---------------- CDNA5 async global->LDS (self-probing guards) --------------
#if defined(__has_builtin)
# if __has_builtin(__builtin_amdgcn_global_load_async_to_lds_b32)
#  define HAS_ASYNC_LDS 1
# endif
#endif

#ifdef HAS_ASYNC_LDS
typedef __attribute__((address_space(1))) int* as1_ip;
typedef __attribute__((address_space(3))) int* as3_ip;
__device__ __forceinline__ void async_g2l_b32(const void* g, void* l) {
  __builtin_amdgcn_global_load_async_to_lds_b32(
      (as1_ip)(uintptr_t)g, (as3_ip)(uintptr_t)l, 0, 0);
}
__device__ __forceinline__ void wait_async0() {
# if __has_builtin(__builtin_amdgcn_s_wait_asynccnt)
  __builtin_amdgcn_s_wait_asynccnt(0);
# else
  asm volatile("s_wait_asynccnt 0" ::: "memory");
# endif
}
#endif

__device__ __forceinline__ void atomic_add_f32(float* p, float v) {
  unsafeAtomicAdd(p, v);    // hardware global_atomic_add_f32 (no CAS loop)
}

typedef __attribute__((ext_vector_type(16))) _Float16 v16h;
typedef __attribute__((ext_vector_type(8)))  float    v8f;

// ---------------------------------- kernels ----------------------------------
__global__ void k_fill(float* __restrict__ p, float v, long n) {
  long i = (long)blockIdx.x * blockDim.x + threadIdx.x;
  if (i < n) p[i] = v;
}

__global__ void k_degree(const int* __restrict__ col, float* __restrict__ deg, int E) {
  int e = blockIdx.x * blockDim.x + threadIdx.x;
  if (e < E) atomic_add_f32(&deg[col[e]], 1.0f);
}

__global__ void k_rsqrt(float* __restrict__ d, int n) {
  int i = blockIdx.x * blockDim.x + threadIdx.x;
  if (i < n) { float x = d[i]; d[i] = (x > 0.0f) ? rsqrtf(x) : 0.0f; }
}

__global__ void k_weights(const int* __restrict__ r, const int* __restrict__ c,
                          const float* __restrict__ dis, float* __restrict__ w, int E) {
  int e = blockIdx.x * blockDim.x + threadIdx.x;
  if (e < E) w[e] = dis[r[e]] * dis[c[e]];
}

// error[mask[i],c] = onehot - y_soft[mask[i],c]; also sigma += |error| (block-reduced)
__global__ void k_build_error(const float* __restrict__ ys, const int* __restrict__ yt,
                              const int* __restrict__ mask, float* __restrict__ err,
                              float* __restrict__ sigma, long mc) {
  __shared__ float red[THREADS];
  float part = 0.0f;
  for (long idx = (long)blockIdx.x * blockDim.x + threadIdx.x; idx < mc;
       idx += (long)gridDim.x * blockDim.x) {
    int  i = (int)(idx / CNUM), c = (int)(idx % CNUM);
    long off = (long)mask[i] * CNUM + c;
    float v = ((yt[i] == c) ? 1.0f : 0.0f) - ys[off];
    err[off] = v;
    part += fabsf(v);
  }
  red[threadIdx.x] = part; __syncthreads();
  for (int s = THREADS / 2; s > 0; s >>= 1) {
    if (threadIdx.x < (unsigned)s) red[threadIdx.x] += red[threadIdx.x + s];
    __syncthreads();
  }
  if (threadIdx.x == 0) atomic_add_f32(sigma, red[0]);
}

__global__ void k_scale_copy(float* __restrict__ dst, const float* __restrict__ src,
                             float s, long n) {
  long i = (long)blockIdx.x * blockDim.x + threadIdx.x;
  if (i < n) dst[i] = s * src[i];
}

__global__ void k_copy(float* __restrict__ dst, const float* __restrict__ src, long n) {
  long i = (long)blockIdx.x * blockDim.x + threadIdx.x;
  if (i < n) dst[i] = src[i];
}

// SpMM edge pass: acc[col] += alpha*w*x[row]. Edge tiles staged via async->LDS.
__global__ void __launch_bounds__(THREADS)
k_spmm_edges(const int* __restrict__ rows, const int* __restrict__ cols,
             const float* __restrict__ w, const float* __restrict__ x,
             float* __restrict__ acc, int E, float alpha) {
  __shared__ int   s_row[TILE];
  __shared__ int   s_col[TILE];
  __shared__ float s_w[TILE];
  const int base = blockIdx.x * TILE;
  const int tid  = threadIdx.x;

#pragma unroll
  for (int t = 0; t < TILE / THREADS; ++t) {
    int li = tid + t * THREADS;
    int e  = base + li;
    if (e < E) {
#ifdef HAS_ASYNC_LDS
      async_g2l_b32(&rows[e], &s_row[li]);
      async_g2l_b32(&cols[e], &s_col[li]);
      async_g2l_b32(&w[e],    &s_w[li]);
#else
      s_row[li] = rows[e]; s_col[li] = cols[e]; s_w[li] = w[e];
#endif
    }
  }
#ifdef HAS_ASYNC_LDS
  wait_async0();
#endif
  __syncthreads();

  const int g = tid >> 3;       // 32 edge-groups per block
  const int j = tid & 7;        // 8 lanes x 5 columns = 40
  for (int s = g; s < TILE; s += 32) {
    int e = base + s;
    if (e >= E) break;
    int   r  = s_row[s];
    int   cc = s_col[s];
    float we = s_w[s] * alpha;
    if (we != 0.0f) {
      const float* xr = x   + (long)r  * CNUM + j * 5;
      float*       ar = acc + (long)cc * CNUM + j * 5;
#pragma unroll
      for (int q = 0; q < 5; ++q) atomic_add_f32(&ar[q], we * xr[q]);
    }
  }
}

// dst = clip(acc, lo, hi); acc = res   (fused reset for next layer)
__global__ void k_combine_reset(float* __restrict__ dst, float* __restrict__ acc,
                                const float* __restrict__ res, float lo, float hi, long n) {
  long i = (long)blockIdx.x * blockDim.x + threadIdx.x;
  if (i < n) {
    float v = acc[i];
    v = fminf(fmaxf(v, lo), hi);
    dst[i] = v;
    acc[i] = res[i];
  }
}

// rowsum[i] = sum_c |x[i,c]| via v_wmma_f32_16x16x32_f16: |A| x ones.
// 16 rows per wave; A laid out per ISA 16-bit A-matrix map; two WMMAs cover c=0..39.
__global__ void __launch_bounds__(THREADS)
k_rowsum_abs_wmma(const float* __restrict__ x, float* __restrict__ rowsum, int n) {
  const unsigned tid  = threadIdx.x;
  const unsigned lane = tid & 31u;
  const unsigned wave = (blockIdx.x * (THREADS / 32)) + (tid >> 5);
  const int  r0  = (int)wave * 16;
  const int  m   = (int)(lane & 15u);
  const int  hi  = (int)(lane >> 4);       // lanes 16..31 hold K 8..15 and 24..31
  const int  row = r0 + m;
  const bool rv  = row < n;
  const float* xr = x + (long)row * CNUM;

  v16h a0, a1;
#pragma unroll
  for (int p = 0; p < 8; ++p) {
    int k  = ((p < 4) ? 2 * p : 16 + 2 * (p - 4)) + hi * 8;   // K for chunk c=0..31
    float v0 = rv ? fabsf(xr[k])     : 0.0f;
    float v1 = rv ? fabsf(xr[k + 1]) : 0.0f;
    a0[2 * p]     = (_Float16)v0;
    a0[2 * p + 1] = (_Float16)v1;
    int k2 = 32 + k;                                          // chunk c=32..39, zero-padded
    float u0 = (rv && (k2     < CNUM)) ? fabsf(xr[k2])     : 0.0f;
    float u1 = (rv && (k2 + 1 < CNUM)) ? fabsf(xr[k2 + 1]) : 0.0f;
    a1[2 * p]     = (_Float16)u0;
    a1[2 * p + 1] = (_Float16)u1;
  }
  v16h b;
#pragma unroll
  for (int p = 0; p < 16; ++p) b[p] = (_Float16)1.0f;

  v8f c = {};
  c = __builtin_amdgcn_wmma_f32_16x16x32_f16(false, a0, false, b, (short)0, c, false, false);
  c = __builtin_amdgcn_wmma_f32_16x16x32_f16(false, a1, false, b, (short)0, c, false, false);

  // D layout: VGPR v holds M=v (lanes 0-15) / M=v+8 (lanes 16-31); take N=0 column.
  if ((lane & 15u) == 0u) {
#pragma unroll
    for (int v = 0; v < 8; ++v) {
      int rr = r0 + v + hi * 8;
      if (rr < n) rowsum[rr] = c[v];
    }
  }
}

__global__ void k_apply_scale(const float* __restrict__ ys, const float* __restrict__ rowsum,
                              const float* __restrict__ sigma, float* __restrict__ y,
                              long n, float invM) {
  long idx = (long)blockIdx.x * blockDim.x + threadIdx.x;
  if (idx < n) {
    long i = idx / CNUM;
    float sc = (sigma[0] * invM) / rowsum[i];
    if (isinf(sc) || sc > 1000.0f) sc = 1.0f;
    y[idx] = ys[idx] + sc * y[idx];
  }
}

__global__ void k_set_onehot(const int* __restrict__ yt, const int* __restrict__ mask,
                             float* __restrict__ y, long mc) {
  long idx = (long)blockIdx.x * blockDim.x + threadIdx.x;
  if (idx < mc) {
    int i = (int)(idx / CNUM), c = (int)(idx % CNUM);
    y[(long)mask[i] * CNUM + c] = (yt[i] == c) ? 1.0f : 0.0f;
  }
}

// --------------------------------- launcher ----------------------------------
static inline unsigned gb(long n) { return (unsigned)((n + THREADS - 1) / THREADS); }

extern "C" void kernel_launch(void* const* d_in, const int* in_sizes, int n_in,
                              void* d_out, int out_size, void* d_ws, size_t ws_size,
                              hipStream_t stream) {
  const float* y_soft = (const float*)d_in[0];
  const int*   y_true = (const int*)d_in[1];
  const int*   mask   = (const int*)d_in[2];
  const int*   e_row  = (const int*)d_in[3];

  const int  N  = in_sizes[0] / CNUM;
  const int  M  = in_sizes[1];
  const int  E  = in_sizes[3] / 2;
  const int* e_col = e_row + E;
  const long NC = (long)N * CNUM;
  const long MC = (long)M * CNUM;

  const int   L1 = 50, L2 = 50;
  const float A1 = 0.9f, A2 = 0.8f;

  // workspace layout (floats): w[E] | dis[N] | res[NC] | cur[NC] | acc[NC] | rowsum[N] | sigma[1]
  float* W      = (float*)d_ws;
  float* dis    = W + E;
  float* res    = dis + N;
  float* cur    = res + NC;
  float* acc    = cur + NC;
  float* rowsum = acc + NC;
  float* sigma  = rowsum + N;

  const unsigned gE  = gb(E);
  const unsigned gN  = gb(N);
  const unsigned gNC = gb(NC);
  const unsigned gMC = gb(MC);
  const unsigned gT  = (unsigned)((E + TILE - 1) / TILE);
  const unsigned gW  = (unsigned)(((long)N + 127) / 128);   // 128 rows per 8-wave block

  // ---- GCN norm: deg -> rsqrt -> per-edge weights ----
  k_fill   <<<gN, THREADS, 0, stream>>>(dis, 0.0f, N);
  k_degree <<<gE, THREADS, 0, stream>>>(e_col, dis, E);
  k_rsqrt  <<<gN, THREADS, 0, stream>>>(dis, N);
  k_weights<<<gE, THREADS, 0, stream>>>(e_row, e_col, dis, W, E);

  // ---- correct phase ----
  k_fill<<<gNC, THREADS, 0, stream>>>(cur, 0.0f, NC);
  k_fill<<<1, THREADS, 0, stream>>>(sigma, 0.0f, 1);
  k_build_error<<<gMC, THREADS, 0, stream>>>(y_soft, y_true, mask, cur, sigma, MC);
  k_scale_copy<<<gNC, THREADS, 0, stream>>>(res, cur, 1.0f - A1, NC);
  k_copy<<<gNC, THREADS, 0, stream>>>(acc, res, NC);
  for (int l = 0; l < L1; ++l) {
    k_spmm_edges<<<gT, THREADS, 0, stream>>>(e_row, e_col, W, cur, acc, E, A1);
    k_combine_reset<<<gNC, THREADS, 0, stream>>>(cur, acc, res, -1.0f, 1.0f, NC);
  }
  k_rowsum_abs_wmma<<<gW, THREADS, 0, stream>>>(cur, rowsum, N);
  k_apply_scale<<<gNC, THREADS, 0, stream>>>(y_soft, rowsum, sigma, cur, NC, 1.0f / (float)M);

  // ---- smooth phase ----
  k_set_onehot<<<gMC, THREADS, 0, stream>>>(y_true, mask, cur, MC);
  k_scale_copy<<<gNC, THREADS, 0, stream>>>(res, cur, 1.0f - A2, NC);
  k_copy<<<gNC, THREADS, 0, stream>>>(acc, res, NC);
  float* outp = (float*)d_out;
  for (int l = 0; l < L2; ++l) {
    k_spmm_edges<<<gT, THREADS, 0, stream>>>(e_row, e_col, W, cur, acc, E, A2);
    float* dst = (l == L2 - 1) ? outp : cur;
    k_combine_reset<<<gNC, THREADS, 0, stream>>>(dst, acc, res, 0.0f, 1.0f, NC);
  }
  (void)n_in; (void)out_size; (void)ws_size;
}
